// SwinAttentionBlock_28673201668288
// MI455X (gfx1250) — matrix operand order repaired and checked by reading, not compiled, not run on hardware
//
#include <hip/hip_runtime.h>

// ---------------------------------------------------------------------------
// Swin window attention, MI455X (gfx1250), wave32 + v_wmma_f32_16x16x32_bf16.
// B=1024 windows, L=64, C=1024, H=16, D=64.
// Pipeline: fp32->bf16 convert -> 3x QKV GEMM (128x128 tile, double-buffered
// LDS, 8 accumulators/wave) -> attention (async-to-LDS staging, WMMA,
// fused softmax + relative-position bias).
// ---------------------------------------------------------------------------

typedef __bf16 v16bf __attribute__((ext_vector_type(16)));
typedef float  v8f   __attribute__((ext_vector_type(8)));

union BFrag { v16bf v; unsigned short u[16]; };

__device__ __forceinline__ unsigned short f2bf(float f) {
  unsigned u = __float_as_uint(f);
  u += 0x7FFFu + ((u >> 16) & 1u);          // round-to-nearest-even
  return (unsigned short)(u >> 16);
}

// A-matrix 16x32 bf16 fragment from row-major LDS (ld elements per row).
// Lane l: m = row0 + (l&15); k-base = k0 + 8*(l>=16).
// Regs 0..7 <- K = kb+0..7, regs 8..15 <- K = kb+16..23 (two b128 loads).
__device__ __forceinline__ v16bf load_a16x32(const unsigned short* A,
                                             int row0, int k0, int ld) {
  const int lane = threadIdx.x & 31;
  const unsigned short* p =
      A + (size_t)(row0 + (lane & 15)) * ld + k0 + ((lane & 16) ? 8 : 0);
  BFrag f;
#pragma unroll
  for (int i = 0; i < 8; ++i) { f.u[i] = p[i]; f.u[8 + i] = p[16 + i]; }
  return f.v;
}

// B-fragment where logical B = M^T, M row-major [n][k] (transposed-staged
// weights, K^T, V^T): regs 0..7 <- K = kb+0..7, regs 8..15 <- K = kb+16..23.
// Lane l: n = n0 + (l&15); k-base = k0 + 8*(l>=16).
__device__ __forceinline__ v16bf load_bT(const unsigned short* M,
                                         int k0, int n0, int ld) {
  const int lane = threadIdx.x & 31;
  const unsigned short* p =
      M + (size_t)(n0 + (lane & 15)) * ld + k0 + ((lane & 16) ? 8 : 0);
  BFrag f;
#pragma unroll
  for (int i = 0; i < 8; ++i) { f.u[i] = p[i]; f.u[8 + i] = p[16 + i]; }
  return f.v;
}

__device__ __forceinline__ v8f wmma_bf16(v16bf a, v16bf b, v8f c) {
  return __builtin_amdgcn_wmma_f32_16x16x32_bf16(false, a, false, b,
                                                 (short)0, c, false, false);
}

// Async global->LDS copy, 16 bytes per lane (ASYNCcnt-tracked DMA path).
// Generic LDS pointers carry the LDS byte offset in the low 32 bits.
__device__ __forceinline__ void async_copy_b128(void* lds, const void* gaddr) {
  asm volatile("global_load_async_to_lds_b128 %0, %1, off"
               :
               : "v"((unsigned)(uintptr_t)lds),
                 "v"((unsigned long long)(uintptr_t)gaddr)
               : "memory");
}
__device__ __forceinline__ void wait_async0() {
  asm volatile("s_wait_asynccnt 0" ::: "memory");
}

// ---------------------------------------------------------------------------
// Kernel 1: fp32 -> bf16 bulk convert (b128 load / b64 store).
// ---------------------------------------------------------------------------
__global__ __launch_bounds__(256) void cvt_bf16(const float4* __restrict__ src,
                                                ushort4* __restrict__ dst,
                                                int n4) {
  int i = blockIdx.x * blockDim.x + threadIdx.x;
  const int stride = gridDim.x * blockDim.x;
  for (; i < n4; i += stride) {
    float4 f = src[i];
    ushort4 u;
    u.x = f2bf(f.x); u.y = f2bf(f.y); u.z = f2bf(f.z); u.w = f2bf(f.w);
    dst[i] = u;
  }
}

// ---------------------------------------------------------------------------
// Kernel 2: one QKV projection.
//   Out[b][h][l][d] = scale * (X[row] . W[:,col] + bias[col])   (bf16)
// 128x128 tile (2 heads), K in 32-deep chunks, double-buffered LDS, W staged
// transposed so B-fragments are contiguous b128 loads.
// Wave w: ntiles {2(w&3), 2(w&3)+1} x mtiles {4(w>>2)..+3} -> 8 accumulators;
// per chunk: 2 B-frags + 4 A-frags feed 8 WMMAs (1.5 b128 ds-loads / WMMA).
// ---------------------------------------------------------------------------
__device__ __forceinline__ void stage_chunk(
    const unsigned short* __restrict__ Xb, const unsigned short* __restrict__ Wb,
    unsigned short* As, unsigned short* Bt, int row0, int col0, int k0,
    int tid) {
  // X chunk [128 x 32], vectorized ushort4.
#pragma unroll
  for (int i = 0; i < 4; ++i) {
    int idx4 = tid + i * 256;                 // 1024 x ushort4
    int r = idx4 >> 3, c4 = (idx4 & 7) << 2;
    *(ushort4*)&As[r * 32 + c4] =
        *(const ushort4*)&Xb[(size_t)(row0 + r) * 1024 + k0 + c4];
  }
  // W chunk [32 x 128] -> transposed Bt[n][k].
#pragma unroll
  for (int i = 0; i < 4; ++i) {
    int idx4 = tid + i * 256;                 // 1024 x ushort4
    int r = idx4 >> 5, c4 = (idx4 & 31) << 2; // r = k, c4 = n
    ushort4 w = *(const ushort4*)&Wb[(size_t)(k0 + r) * 1024 + col0 + c4];
    Bt[(c4 + 0) * 32 + r] = w.x;
    Bt[(c4 + 1) * 32 + r] = w.y;
    Bt[(c4 + 2) * 32 + r] = w.z;
    Bt[(c4 + 3) * 32 + r] = w.w;
  }
}

__global__ __launch_bounds__(256) void qkv_gemm(
    const float* __restrict__ bias, const unsigned short* __restrict__ Xb,
    const unsigned short* __restrict__ Wb, unsigned short* __restrict__ Out,
    float scale) {
  __shared__ unsigned short As[2][128 * 32];  // 2 x 8 KB
  __shared__ unsigned short Bt[2][128 * 32];  // 2 x 8 KB (transposed W)

  const int tid = threadIdx.x;
  const int lane = tid & 31;
  const int wv = tid >> 5;
  const int row0 = blockIdx.x * 128;
  const int col0 = blockIdx.y * 128;          // spans heads 2y, 2y+1
  const int ntb = (wv & 3) * 2;               // 2 consecutive N-tiles
  const int mtb = (wv >> 2) * 4;              // 4 consecutive M-tiles

  v8f acc[4][2] = {};

  stage_chunk(Xb, Wb, As[0], Bt[0], row0, col0, 0, tid);

  for (int kc = 0; kc < 32; ++kc) {
    __syncthreads();
    if (kc + 1 < 32) {
      if (kc + 2 < 32) {  // prefetch chunk kc+2 (global_prefetch_b8)
        __builtin_prefetch(&Xb[(size_t)(row0 + (tid >> 1)) * 1024 +
                               (kc + 2) * 32 + ((tid & 1) << 4)], 0, 3);
        if (tid < 64)
          __builtin_prefetch(&Wb[(size_t)((kc + 2) * 32 + (tid & 31)) * 1024 +
                                 col0 + ((tid >> 5) << 6)], 0, 3);
      }
      stage_chunk(Xb, Wb, As[(kc + 1) & 1], Bt[(kc + 1) & 1], row0, col0,
                  (kc + 1) * 32, tid);
    }
    const unsigned short* as = As[kc & 1];
    const unsigned short* bt = Bt[kc & 1];
    v16bf bfr[2];
    bfr[0] = load_bT(bt, 0, (ntb + 0) * 16, 32);
    bfr[1] = load_bT(bt, 0, (ntb + 1) * 16, 32);
#pragma unroll
    for (int i = 0; i < 4; ++i) {
      v16bf af = load_a16x32(as, (mtb + i) * 16, 0, 32);
      acc[i][0] = wmma_bf16(af, bfr[0], acc[i][0]);
      acc[i][1] = wmma_bf16(af, bfr[1], acc[i][1]);
    }
  }

  // Epilogue: + bias, * scale, store bf16 at [b][h][l][d].
  const int nl = lane & 15;
  const int hi = (lane & 16) ? 8 : 0;
#pragma unroll
  for (int n = 0; n < 2; ++n) {
    const int col = col0 + (ntb + n) * 16 + nl;
    const int h = col >> 6, d = col & 63;
    const float bv = bias[col];
#pragma unroll
    for (int i = 0; i < 4; ++i) {
#pragma unroll
      for (int j = 0; j < 8; ++j) {
        int r = row0 + (mtb + i) * 16 + hi + j;  // global row = b*64 + l
        float v = (acc[i][n][j] + bv) * scale;
        int b = r >> 6, l = r & 63;
        Out[((size_t)(b * 16 + h) * 64 + l) * 64 + d] = f2bf(v);
      }
    }
  }
}

// ---------------------------------------------------------------------------
// Kernel 3: attention for one (b, h). 4 waves (128 threads).
//   S = Qh @ Kh^T (scale pre-folded into Q) + rel_bias; P = softmax(S);
//   O = P @ Vh  -> d_out[b][l][h*64+d]  (fp32)
// Q/K staged via async-to-LDS DMA; V staged transposed for b128 B-frags.
// ---------------------------------------------------------------------------
__global__ __launch_bounds__(128) void attn(
    const unsigned short* __restrict__ Qb, const unsigned short* __restrict__ Kb,
    const unsigned short* __restrict__ Vb, const float* __restrict__ bias_table,
    float* __restrict__ out) {
  __shared__ unsigned short Qs[4096], Ks[4096], Vt[4096], Ps[4096];
  __shared__ float Sc[64 * 68];
  __shared__ float biasc[127];

  const int h = blockIdx.x, b = blockIdx.y;
  const int tid = threadIdx.x;
  const int lane = tid & 31;
  const int mt = tid >> 5;  // wave id = M-tile

  const size_t base = ((size_t)b * 16 + h) * 4096;

  // Q, K: pure bf16 copies -> async DMA straight into LDS (ASYNCcnt).
#pragma unroll
  for (int c = tid; c < 512; c += 128) {  // 512 x 16B per matrix
    async_copy_b128(&Qs[c * 8], Qb + base + c * 8);
    async_copy_b128(&Ks[c * 8], Kb + base + c * 8);
  }
  // V: transpose on the way into LDS -> Vt[d][l].
  for (int c = tid; c < 1024; c += 128) {
    ushort4 w = *(const ushort4*)&Vb[base + c * 4];
    int l = (c * 4) >> 6, d = (c * 4) & 63;
    Vt[(d + 0) * 64 + l] = w.x;
    Vt[(d + 1) * 64 + l] = w.y;
    Vt[(d + 2) * 64 + l] = w.z;
    Vt[(d + 3) * 64 + l] = w.w;
  }
  if (tid < 127) biasc[tid] = bias_table[tid * 16 + h];  // column h of table
  wait_async0();
  __syncthreads();

  // S = Q @ K^T : wave mt owns rows mt*16..+15, all 64 cols.
  v8f sacc[4] = {};
#pragma unroll
  for (int kc = 0; kc < 2; ++kc) {
    v16bf af = load_a16x32(Qs, mt * 16, kc * 32, 64);
#pragma unroll
    for (int n = 0; n < 4; ++n) {
      v16bf bf = load_bT(Ks, kc * 32, n * 16, 64);  // B = K^T
      sacc[n] = wmma_bf16(af, bf, sacc[n]);
    }
  }
  const int nl = lane & 15;
  const int hi = (lane & 16) ? 8 : 0;
#pragma unroll
  for (int n = 0; n < 4; ++n)
#pragma unroll
    for (int j = 0; j < 8; ++j)
      Sc[(size_t)(mt * 16 + hi + j) * 68 + n * 16 + nl] = sacc[n][j];
  __syncthreads();

  // Softmax over k with Swin relative-position bias: idx(i,j) = i - j + 63.
  if (tid < 64) {
    const int row = tid;
    float* s = &Sc[(size_t)row * 68];
    float m = -1e30f;
#pragma unroll 8
    for (int j = 0; j < 64; ++j) {
      float v = s[j] + biasc[row - j + 63];
      s[j] = v;
      m = fmaxf(m, v);
    }
    float sum = 0.f;
#pragma unroll 8
    for (int j = 0; j < 64; ++j) {
      float e = __expf(s[j] - m);
      s[j] = e;
      sum += e;
    }
    const float inv = 1.f / sum;
#pragma unroll 8
    for (int j = 0; j < 64; ++j) Ps[row * 64 + j] = f2bf(s[j] * inv);
  }
  __syncthreads();

  // O = P @ V  (B-frags from transposed Vt -> contiguous b128 loads).
  v8f oacc[4] = {};
#pragma unroll
  for (int kc = 0; kc < 2; ++kc) {
    v16bf af = load_a16x32(Ps, mt * 16, kc * 32, 64);
#pragma unroll
    for (int n = 0; n < 4; ++n) {
      v16bf bf = load_bT(Vt, kc * 32, n * 16, 64);
      oacc[n] = wmma_bf16(af, bf, oacc[n]);
    }
  }
#pragma unroll
  for (int n = 0; n < 4; ++n)
#pragma unroll
    for (int j = 0; j < 8; ++j) {
      int l = mt * 16 + hi + j;
      int d = n * 16 + nl;
      out[((size_t)b * 64 + l) * 1024 + h * 64 + d] = oacc[n][j];
    }
}

// ---------------------------------------------------------------------------
// Launcher. Workspace layout (ushort elements):
//   Xb @ 0        : 65536*1024   = 128 MB
//   Wb @ 67108864 : 3*1024*1024  =   6 MB   (q, k, v)
//   Q  @ 70254592 : 1024*16*64*64 = 128 MB; K, V follow (total ~518 MB).
// ---------------------------------------------------------------------------
extern "C" void kernel_launch(void* const* d_in, const int* in_sizes, int n_in,
                              void* d_out, int out_size, void* d_ws,
                              size_t ws_size, hipStream_t stream) {
  const float* X  = (const float*)d_in[0];
  const float* Wq = (const float*)d_in[1];
  const float* bq = (const float*)d_in[2];
  const float* Wk = (const float*)d_in[3];
  const float* bk = (const float*)d_in[4];
  const float* Wv = (const float*)d_in[5];
  const float* bv = (const float*)d_in[6];
  const float* bt = (const float*)d_in[7];
  // d_in[8] (rel_index) is the closed form i - j + 63; used analytically.

  unsigned short* ws = (unsigned short*)d_ws;
  const size_t XB = 0;
  const size_t WB = 67108864ull;
  const size_t QO = WB + 3145728ull;
  const size_t KO = QO + 67108864ull;
  const size_t VO = KO + 67108864ull;

  // 1) fp32 -> bf16 conversions (bf16 X is 128MB -> L2-resident for re-reads).
  cvt_bf16<<<4096, 256, 0, stream>>>((const float4*)X, (ushort4*)(ws + XB),
                                     67108864 / 4);
  cvt_bf16<<<512, 256, 0, stream>>>((const float4*)Wq, (ushort4*)(ws + WB),
                                    1048576 / 4);
  cvt_bf16<<<512, 256, 0, stream>>>((const float4*)Wk,
                                    (ushort4*)(ws + WB + 1048576), 1048576 / 4);
  cvt_bf16<<<512, 256, 0, stream>>>((const float4*)Wv,
                                    (ushort4*)(ws + WB + 2097152), 1048576 / 4);

  // 2) QKV projections (1/sqrt(64) folded into Q), 128x128 tiles.
  dim3 gg(512, 8);
  qkv_gemm<<<gg, 256, 0, stream>>>(bq, ws + XB, ws + WB, ws + QO, 0.125f);
  qkv_gemm<<<gg, 256, 0, stream>>>(bk, ws + XB, ws + WB + 1048576, ws + KO,
                                   1.0f);
  qkv_gemm<<<gg, 256, 0, stream>>>(bv, ws + XB, ws + WB + 2097152, ws + VO,
                                   1.0f);

  // 3) Windowed attention, one block per (head, window).
  attn<<<dim3(16, 1024), 128, 0, stream>>>(ws + QO, ws + KO, ws + VO, bt,
                                           (float*)d_out);
}
